// GRU_D_56100862820848
// MI455X (gfx1250) — matrix operand order repaired and checked by reading, not compiled, not run on hardware
//
#include <hip/hip_runtime.h>
#include <hip/hip_bf16.h>
#include <math.h>

#define Bsz 256
#define Tsz 256
#define Isz 128
#define Hsz 512
#define Osz 128
#define KTOT 768   /* I + H + I */

typedef __attribute__((ext_vector_type(16))) __bf16 v16bf;
typedef __attribute__((ext_vector_type(8)))  __bf16 v8bf;
typedef __attribute__((ext_vector_type(8)))  float  v8f;

// ---------------------------------------------------------------------------
// Fragment helpers (CDNA5 wave32 WMMA 16x16x32 bf16 layouts, ISA 05_wmma §7.12.2)
// A (16x32, row M = lane&15, half = lane>>4): VGPR0-3 K=8*half+[0,8), VGPR4-7 K=16+8*half+[0,8)
// B (32x16, col N = lane&15, half = lane>>4): lane holds K = 8*half+[0,8), 16+8*half+[0,8) (pre-packed)
// C (16x16 f32): VGPR r -> M = r + 8*half, N = lane&15
// ---------------------------------------------------------------------------
__device__ __forceinline__ v16bf load2x8(const __bf16* p0, const __bf16* p1) {
    v16bf a;
    *(v8bf*)&a       = *(const v8bf*)p0;
    *((v8bf*)&a + 1) = *(const v8bf*)p1;
    return a;
}

// Read one B fragment from LDS-staged chunk (ds_load_b128 x2).
__device__ __forceinline__ v16bf load_bfrag_lds(const __bf16* sB, int ngl, int lane) {
    const __bf16* p = sB + ngl * 512 + lane * 16;
    return load2x8(p, p + 8);
}

// Direct-from-global B fragment (used by decoder GEMM).
__device__ __forceinline__ v16bf load_bfrag(const __bf16* __restrict__ pack,
                                            int kc, int ng, int NGtot, int lane) {
    const __bf16* p = pack + ((size_t)(kc * NGtot + ng)) * 512 + lane * 16;
    return load2x8(p, p + 8);
}

// Async global->LDS stage of one 8 KB B chunk (8 n-groups x 512 bf16, contiguous
// in the packed layout). 256 threads x 2 async b128 transfers. Tracked by
// ASYNCcnt; generic-pointer low 32 bits == LDS offset (ISA aperture mapping).
__device__ __forceinline__ void stage_b_chunk(const __bf16* __restrict__ gsrc,
                                              __bf16* sdst, int tid) {
    unsigned lbase = (unsigned)(uintptr_t)sdst;
#pragma unroll
    for (int rturn = 0; rturn < 2; ++rturn) {
        unsigned o = (unsigned)(tid + rturn * 256) * 16u;
        unsigned long long ga = (unsigned long long)(uintptr_t)((const char*)gsrc + o);
        asm volatile("global_load_async_to_lds_b128 %0, %1, off"
                     :: "v"(lbase + o), "v"(ga) : "memory");
    }
}

__device__ __forceinline__ void wait_async_and_sync() {
    asm volatile("s_wait_asynccnt 0" ::: "memory");
    __syncthreads();
}

__device__ __forceinline__ float sigmoidf_(float x) {
    return 1.0f / (1.0f + __expf(-x));
}

// ---------------------------------------------------------------------------
// Weight packer: src [Ks x Ns] fp32 row-major -> bf16 fragment-native layout.
// Packed element addr = ((kc*NGtot + ng)*512) + (half*16 + n)*16 + kr, kr = (kk&7) + 8*(kk>>4)
// ---------------------------------------------------------------------------
__global__ void pack_b_kernel(const float* __restrict__ src, __bf16* __restrict__ dst,
                              int Ks, int Ns, int NGtot, int kc0, int ng0) {
    int idx = blockIdx.x * blockDim.x + threadIdx.x;
    int total = Ks * Ns;
    if (idx >= total) return;
    int k = idx / Ns;
    int n = idx - k * Ns;
    int kc   = kc0 + (k >> 5);
    int kk   = k & 31;
    int half = (kk >> 3) & 1;
    int kr   = (kk & 7) + ((kk >> 4) << 3);
    int ng   = ng0 + (n >> 4);
    int nl   = n & 15;
    size_t off = ((size_t)(kc * NGtot + ng)) * 512 + (half * 16 + nl) * 16 + kr;
    dst[off] = (__bf16)src[(size_t)k * Ns + n];
}

// ---------------------------------------------------------------------------
// Imputation / preprocessing: ximp, (1-m), m  -> bf16
// ---------------------------------------------------------------------------
__global__ void prep_kernel(const float* __restrict__ x, const float* __restrict__ delta,
                            const float* __restrict__ m, const float* __restrict__ xf,
                            const float* __restrict__ wgx, const float* __restrict__ bgx,
                            __bf16* __restrict__ ximp, __bf16* __restrict__ omm,
                            __bf16* __restrict__ mbf) {
    size_t i = (size_t)blockIdx.x * blockDim.x + threadIdx.x;
    if (i >= (size_t)Bsz * Tsz * Isz) return;
    int ii = (int)(i & (Isz - 1));
    float mv = m[i];
    float dx = delta[i] * wgx[ii] + bgx[ii];
    float xr = dx * xf[i] + (1.0f - dx) * 0.001f;
    float xv = (mv > 0.0f) ? xr : x[i];
    ximp[i] = (__bf16)xv;
    omm[i]  = (__bf16)(1.0f - mv);
    mbf[i]  = (__bf16)mv;
}

__global__ void zero_kernel(float* __restrict__ p, int n) {
    int i = blockIdx.x * blockDim.x + threadIdx.x;
    if (i < n) p[i] = 0.0f;
}

// ---------------------------------------------------------------------------
// gamma_h = exp(-relu(m @ W_gamma_h + b_gamma_h)) : [B*T,128] x [128,512] -> bf16
// Block = 256 thr (8 waves, 4M x 2N), block tile 64x128, wave tile 16x64.
// B chunk staged in LDS via async copies.
// ---------------------------------------------------------------------------
__global__ void __launch_bounds__(256) gamma_gemm_kernel(
        const __bf16* __restrict__ Mbf, const __bf16* __restrict__ Wgh,
        const float* __restrict__ bgh, __bf16* __restrict__ G) {
    const int NGtot = Hsz / 16;  // 32
    __shared__ __bf16 sB[8 * 512];
    int tid  = threadIdx.x;
    int lane = tid & 31;
    int wave = tid >> 5;
    int wm = wave & 3, wn = wave >> 2;
    int m0 = blockIdx.x * 64 + wm * 16;
    int n0 = blockIdx.y * 128 + wn * 64;
    int ng0 = blockIdx.y * 8;  // 8 n-groups per block
    int row = m0 + (lane & 15);
    int half = lane >> 4;
    v8f acc[4] = {};
    for (int kc = 0; kc < Isz / 32; ++kc) {
        stage_b_chunk(Wgh + ((size_t)(kc * NGtot + ng0)) * 512, sB, tid);
        const __bf16* p = Mbf + (size_t)row * Isz + kc * 32 + 8 * half;
        v16bf a = load2x8(p, p + 16);
        wait_async_and_sync();
#pragma unroll
        for (int j = 0; j < 4; ++j) {
            v16bf b = load_bfrag_lds(sB, wn * 4 + j, lane);
            acc[j] = __builtin_amdgcn_wmma_f32_16x16x32_bf16(
                false, a, false, b, (short)0, acc[j], false, false);
        }
        __syncthreads();
    }
    int nl = lane & 15;
#pragma unroll
    for (int j = 0; j < 4; ++j) {
        int nn = n0 + j * 16 + nl;
#pragma unroll
        for (int r = 0; r < 8; ++r) {
            int mm = m0 + r + 8 * half;
            float v = acc[j][r] + bgh[nn];
            v = __expf(-fmaxf(v, 0.0f));
            G[(size_t)mm * Hsz + nn] = (__bf16)v;
        }
    }
}

// ---------------------------------------------------------------------------
// Recurrence phase A (per step t): [z|r] = sigmoid([x_t, g*h, 1-m_t] @ Wzr + b)
// C is [256 x 1024]; grid (4, 8). B staged in LDS (async).
// ---------------------------------------------------------------------------
__global__ void __launch_bounds__(256) step_a_kernel(
        const __bf16* __restrict__ ximp, const __bf16* __restrict__ omm,
        const __bf16* __restrict__ G, const float* __restrict__ hcur,
        const __bf16* __restrict__ Wzr, const float* __restrict__ b_z,
        const float* __restrict__ b_r, float* __restrict__ zbuf,
        float* __restrict__ rbuf, int t) {
    const int NGtot = 1024 / 16;  // 64
    __shared__ __bf16 sB[8 * 512];
    int tid  = threadIdx.x;
    int lane = tid & 31;
    int wave = tid >> 5;
    int wm = wave & 3, wn = wave >> 2;
    int m0 = blockIdx.x * 64 + wm * 16;
    int n0 = blockIdx.y * 128 + wn * 64;
    int ng0 = blockIdx.y * 8;
    int row = m0 + (lane & 15);
    int half = lane >> 4;
    const __bf16* xrow = ximp + ((size_t)row * Tsz + t) * Isz;
    const __bf16* orow = omm  + ((size_t)row * Tsz + t) * Isz;
    const __bf16* grow = G    + ((size_t)row * Tsz + t) * Hsz;
    const float*  hrow = hcur + (size_t)row * Hsz;
    v8f acc[4] = {};
    for (int kc = 0; kc < KTOT / 32; ++kc) {
        stage_b_chunk(Wzr + ((size_t)(kc * NGtot + ng0)) * 512, sB, tid);
        v16bf a;
        if (kc < 4) {
            const __bf16* p = xrow + kc * 32 + 8 * half;
            a = load2x8(p, p + 16);
        } else if (kc < 20) {
            int hb = (kc - 4) * 32 + 8 * half;
#pragma unroll
            for (int i = 0; i < 8; ++i) {
                a[i]     = (__bf16)((float)grow[hb + i]      * hrow[hb + i]);
                a[i + 8] = (__bf16)((float)grow[hb + 16 + i] * hrow[hb + 16 + i]);
            }
        } else {
            const __bf16* p = orow + (kc - 20) * 32 + 8 * half;
            a = load2x8(p, p + 16);
        }
        wait_async_and_sync();
#pragma unroll
        for (int j = 0; j < 4; ++j) {
            v16bf b = load_bfrag_lds(sB, wn * 4 + j, lane);
            acc[j] = __builtin_amdgcn_wmma_f32_16x16x32_bf16(
                false, a, false, b, (short)0, acc[j], false, false);
        }
        __syncthreads();
    }
    int nl = lane & 15;
#pragma unroll
    for (int j = 0; j < 4; ++j) {
        int nn = n0 + j * 16 + nl;
#pragma unroll
        for (int r = 0; r < 8; ++r) {
            int mm = m0 + r + 8 * half;
            float v = acc[j][r];
            if (nn < Hsz) {
                zbuf[(size_t)mm * Hsz + nn] = sigmoidf_(v + b_z[nn]);
            } else {
                rbuf[(size_t)mm * Hsz + (nn - Hsz)] = sigmoidf_(v + b_r[nn - Hsz]);
            }
        }
    }
}

// ---------------------------------------------------------------------------
// Recurrence phase B: h_tilde = tanh([x_t, g*h*r, 1-m_t] @ Wc + b)
// h_next = (1-z)*(g*h) + z*h_tilde.  C is [256 x 512]; grid (4, 4). B staged.
// ---------------------------------------------------------------------------
__global__ void __launch_bounds__(256) step_b_kernel(
        const __bf16* __restrict__ ximp, const __bf16* __restrict__ omm,
        const __bf16* __restrict__ G, const float* __restrict__ hcur,
        const float* __restrict__ rbuf, const float* __restrict__ zbuf,
        const __bf16* __restrict__ Wc, const float* __restrict__ b_c,
        float* __restrict__ hnext, int t) {
    const int NGtot = Hsz / 16;  // 32
    __shared__ __bf16 sB[8 * 512];
    int tid  = threadIdx.x;
    int lane = tid & 31;
    int wave = tid >> 5;
    int wm = wave & 3, wn = wave >> 2;
    int m0 = blockIdx.x * 64 + wm * 16;
    int n0 = blockIdx.y * 128 + wn * 64;
    int ng0 = blockIdx.y * 8;
    int row = m0 + (lane & 15);
    int half = lane >> 4;
    const __bf16* xrow = ximp + ((size_t)row * Tsz + t) * Isz;
    const __bf16* orow = omm  + ((size_t)row * Tsz + t) * Isz;
    const __bf16* grow = G    + ((size_t)row * Tsz + t) * Hsz;
    const float*  hrow = hcur + (size_t)row * Hsz;
    const float*  rrow = rbuf + (size_t)row * Hsz;
    v8f acc[4] = {};
    for (int kc = 0; kc < KTOT / 32; ++kc) {
        stage_b_chunk(Wc + ((size_t)(kc * NGtot + ng0)) * 512, sB, tid);
        v16bf a;
        if (kc < 4) {
            const __bf16* p = xrow + kc * 32 + 8 * half;
            a = load2x8(p, p + 16);
        } else if (kc < 20) {
            int hb = (kc - 4) * 32 + 8 * half;
#pragma unroll
            for (int i = 0; i < 8; ++i) {
                a[i]     = (__bf16)((float)grow[hb + i]      * hrow[hb + i]      * rrow[hb + i]);
                a[i + 8] = (__bf16)((float)grow[hb + 16 + i] * hrow[hb + 16 + i] * rrow[hb + 16 + i]);
            }
        } else {
            const __bf16* p = orow + (kc - 20) * 32 + 8 * half;
            a = load2x8(p, p + 16);
        }
        wait_async_and_sync();
#pragma unroll
        for (int j = 0; j < 4; ++j) {
            v16bf b = load_bfrag_lds(sB, wn * 4 + j, lane);
            acc[j] = __builtin_amdgcn_wmma_f32_16x16x32_bf16(
                false, a, false, b, (short)0, acc[j], false, false);
        }
        __syncthreads();
    }
    int nl = lane & 15;
#pragma unroll
    for (int j = 0; j < 4; ++j) {
        int nn = n0 + j * 16 + nl;
#pragma unroll
        for (int r = 0; r < 8; ++r) {
            int mm = m0 + r + 8 * half;
            float ht = tanhf(acc[j][r] + b_c[nn]);
            float gv = (float)G[((size_t)mm * Tsz + t) * Hsz + nn];
            float hd = gv * hcur[(size_t)mm * Hsz + nn];
            float z  = zbuf[(size_t)mm * Hsz + nn];
            hnext[(size_t)mm * Hsz + nn] = (1.0f - z) * hd + z * ht;
        }
    }
}

// ---------------------------------------------------------------------------
// BatchNorm (eval): h_bn = h / sqrt(1+eps) * gamma + beta  -> output h_t section
// ---------------------------------------------------------------------------
__global__ void bnout_kernel(const float* __restrict__ h, const float* __restrict__ gamma,
                             const float* __restrict__ beta, float* __restrict__ out_h) {
    int i = blockIdx.x * blockDim.x + threadIdx.x;
    if (i >= Bsz * Hsz) return;
    int n = i & (Hsz - 1);
    const float inv = 0.99999500003749968752f;  // 1/sqrt(1+1e-5)
    out_h[i] = h[i] * inv * gamma[n] + beta[n];
}

// ---------------------------------------------------------------------------
// Decoder GEMM: logits[256,128] = h_bn[256,512] @ dec_W + dec_b ; grid (4, 1)
// ---------------------------------------------------------------------------
__global__ void __launch_bounds__(256) dec_gemm_kernel(
        const float* __restrict__ hbn, const __bf16* __restrict__ Wd,
        const float* __restrict__ dbias, float* __restrict__ logits) {
    const int NGtot = Osz / 16;  // 8
    int lane = threadIdx.x & 31;
    int wave = threadIdx.x >> 5;
    int wm = wave & 3, wn = wave >> 2;
    int m0 = blockIdx.x * 64 + wm * 16;
    int n0 = wn * 64;
    int row = m0 + (lane & 15);
    int half = lane >> 4;
    v8f acc[4] = {};
    for (int kc = 0; kc < Hsz / 32; ++kc) {
        const float* p = hbn + (size_t)row * Hsz + kc * 32 + 8 * half;
        v16bf a;
#pragma unroll
        for (int i = 0; i < 8; ++i) {
            a[i]     = (__bf16)p[i];
            a[i + 8] = (__bf16)p[16 + i];
        }
#pragma unroll
        for (int j = 0; j < 4; ++j) {
            v16bf b = load_bfrag(Wd, kc, (n0 >> 4) + j, NGtot, lane);
            acc[j] = __builtin_amdgcn_wmma_f32_16x16x32_bf16(
                false, a, false, b, (short)0, acc[j], false, false);
        }
    }
    int nl = lane & 15;
#pragma unroll
    for (int j = 0; j < 4; ++j) {
        int nn = n0 + j * 16 + nl;
#pragma unroll
        for (int r = 0; r < 8; ++r) {
            int mm = m0 + r + 8 * half;
            logits[(size_t)mm * Osz + nn] = acc[j][r] + dbias[nn];
        }
    }
}

// ---------------------------------------------------------------------------
// Row-wise log_softmax over 128 logits; one block (128 thr) per row.
// ---------------------------------------------------------------------------
__global__ void __launch_bounds__(128) logsoftmax_kernel(const float* __restrict__ logits,
                                                         float* __restrict__ out) {
    __shared__ float sdata[128];
    int b = blockIdx.x;
    int tid = threadIdx.x;
    float v = logits[(size_t)b * Osz + tid];
    sdata[tid] = v;
    __syncthreads();
    for (int s = 64; s > 0; s >>= 1) {
        if (tid < s) sdata[tid] = fmaxf(sdata[tid], sdata[tid + s]);
        __syncthreads();
    }
    float mx = sdata[0];
    __syncthreads();
    sdata[tid] = __expf(v - mx);
    __syncthreads();
    for (int s = 64; s > 0; s >>= 1) {
        if (tid < s) sdata[tid] += sdata[tid + s];
        __syncthreads();
    }
    float lse = __logf(sdata[0]);
    out[(size_t)b * Osz + tid] = v - mx - lse;
}

// ---------------------------------------------------------------------------
extern "C" void kernel_launch(void* const* d_in, const int* in_sizes, int n_in,
                              void* d_out, int out_size, void* d_ws, size_t ws_size,
                              hipStream_t stream) {
    (void)in_sizes; (void)n_in; (void)out_size; (void)ws_size;
    const float* x         = (const float*)d_in[0];
    const float* delta     = (const float*)d_in[1];
    const float* m         = (const float*)d_in[2];
    const float* x_forward = (const float*)d_in[3];
    const float* W_r       = (const float*)d_in[4];
    const float* U_r       = (const float*)d_in[5];
    const float* V_r       = (const float*)d_in[6];
    const float* b_r       = (const float*)d_in[7];
    const float* W_z       = (const float*)d_in[8];
    const float* U_z       = (const float*)d_in[9];
    const float* V_z       = (const float*)d_in[10];
    const float* b_z       = (const float*)d_in[11];
    const float* W_c       = (const float*)d_in[12];
    const float* U_c       = (const float*)d_in[13];
    const float* V_c       = (const float*)d_in[14];
    const float* b_c       = (const float*)d_in[15];
    const float* W_gamma_x = (const float*)d_in[16];
    const float* b_gamma_x = (const float*)d_in[17];
    const float* W_gamma_h = (const float*)d_in[18];
    const float* b_gamma_h = (const float*)d_in[19];
    const float* dec_W     = (const float*)d_in[20];
    const float* dec_b     = (const float*)d_in[21];
    const float* bn_gamma  = (const float*)d_in[22];
    const float* bn_beta   = (const float*)d_in[23];

    float* out_logp = (float*)d_out;                 // [B, O]
    float* out_h    = (float*)d_out + Bsz * Osz;     // [B, H]

    size_t off = 0;
    auto alloc = [&](size_t bytes) -> void* {
        void* p = (char*)d_ws + off;
        off += (bytes + 255) & ~(size_t)255;
        return p;
    };
    __bf16* ximp = (__bf16*)alloc((size_t)Bsz * Tsz * Isz * 2);
    __bf16* omm  = (__bf16*)alloc((size_t)Bsz * Tsz * Isz * 2);
    __bf16* mbf  = (__bf16*)alloc((size_t)Bsz * Tsz * Isz * 2);
    __bf16* G    = (__bf16*)alloc((size_t)Bsz * Tsz * Hsz * 2);
    __bf16* Wzr  = (__bf16*)alloc((size_t)KTOT * 1024 * 2);
    __bf16* Wc   = (__bf16*)alloc((size_t)KTOT * Hsz * 2);
    __bf16* Wgh  = (__bf16*)alloc((size_t)Isz * Hsz * 2);
    __bf16* Wd   = (__bf16*)alloc((size_t)Hsz * Osz * 2);
    float* h0    = (float*)alloc((size_t)Bsz * Hsz * 4);
    float* h1    = (float*)alloc((size_t)Bsz * Hsz * 4);
    float* zb    = (float*)alloc((size_t)Bsz * Hsz * 4);
    float* rb    = (float*)alloc((size_t)Bsz * Hsz * 4);
    float* logits = (float*)alloc((size_t)Bsz * Osz * 4);

    // 1) preprocessing (imputation + bf16 casts)
    {
        size_t total = (size_t)Bsz * Tsz * Isz;
        prep_kernel<<<(unsigned)((total + 255) / 256), 256, 0, stream>>>(
            x, delta, m, x_forward, W_gamma_x, b_gamma_x, ximp, omm, mbf);
    }

    // 2) pack weights into WMMA-fragment-native bf16 layout
    auto packB = [&](const float* src, __bf16* dst, int Ks, int Ns, int NGtot,
                     int kc0, int ng0) {
        int total = Ks * Ns;
        pack_b_kernel<<<(total + 255) / 256, 256, 0, stream>>>(src, dst, Ks, Ns,
                                                               NGtot, kc0, ng0);
    };
    // Wzr: [W_z;U_z;V_z | W_r;U_r;V_r] -> [768 x 1024]
    packB(W_z, Wzr, 128, 512, 64, 0, 0);
    packB(U_z, Wzr, 512, 512, 64, 4, 0);
    packB(V_z, Wzr, 128, 512, 64, 20, 0);
    packB(W_r, Wzr, 128, 512, 64, 0, 32);
    packB(U_r, Wzr, 512, 512, 64, 4, 32);
    packB(V_r, Wzr, 128, 512, 64, 20, 32);
    // Wc: [W;U;V] -> [768 x 512]
    packB(W_c, Wc, 128, 512, 32, 0, 0);
    packB(U_c, Wc, 512, 512, 32, 4, 0);
    packB(V_c, Wc, 128, 512, 32, 20, 0);
    packB(W_gamma_h, Wgh, 128, 512, 32, 0, 0);
    packB(dec_W, Wd, 512, 128, 8, 0, 0);

    // 3) gamma_h for all (b,t): [65536 x 128] @ [128 x 512]
    gamma_gemm_kernel<<<dim3((Bsz * Tsz) / 64, Hsz / 128), 256, 0, stream>>>(
        mbf, Wgh, b_gamma_h, G);

    // 4) h0 = 0
    zero_kernel<<<(Bsz * Hsz + 255) / 256, 256, 0, stream>>>(h0, Bsz * Hsz);

    // 5) sequential scan over T steps (2 WMMA GEMM phases per step)
    float* hb[2] = {h0, h1};
    int p = 0;
    for (int t = 0; t < Tsz; ++t) {
        step_a_kernel<<<dim3(Bsz / 64, 1024 / 128), 256, 0, stream>>>(
            ximp, omm, G, hb[p], Wzr, b_z, b_r, zb, rb, t);
        step_b_kernel<<<dim3(Bsz / 64, Hsz / 128), 256, 0, stream>>>(
            ximp, omm, G, hb[p], rb, zb, Wc, b_c, hb[p ^ 1], t);
        p ^= 1;
    }
    float* hfin = hb[p];  // p is back to 0 after 256 steps

    // 6) BatchNorm (eval) -> h_t output section
    bnout_kernel<<<(Bsz * Hsz + 255) / 256, 256, 0, stream>>>(hfin, bn_gamma,
                                                              bn_beta, out_h);
    // 7) decoder GEMM + bias
    dec_gemm_kernel<<<dim3(Bsz / 64, 1), 256, 0, stream>>>(out_h, Wd, dec_b, logits);
    // 8) log_softmax -> output section
    logsoftmax_kernel<<<Bsz, 128, 0, stream>>>(logits, out_logp);
}
